// DisentangledSelfAttention_79766132621336
// MI455X (gfx1250) — matrix operand order repaired and checked by reading, not compiled, not run on hardware
//
#include <hip/hip_runtime.h>
#include <hip/hip_bf16.h>
#include <cmath>
#include <cstdint>

typedef __attribute__((ext_vector_type(16))) __bf16        v16bf;
typedef __attribute__((ext_vector_type(8)))  float         v8f;
typedef __attribute__((ext_vector_type(8)))  unsigned int  v8u;

// ---------- helpers ----------
__device__ __forceinline__ unsigned int pack_bf16(float a, float b) {
  unsigned int ua = __float_as_uint(a);
  unsigned int ub = __float_as_uint(b);
  ua += 0x7fffu + ((ua >> 16) & 1u);   // round-to-nearest-even
  ub += 0x7fffu + ((ub >> 16) & 1u);
  return (ua >> 16) | (ub & 0xffff0000u);
}
__device__ __forceinline__ unsigned short f_to_bf16(float a) {
  unsigned int ua = __float_as_uint(a);
  ua += 0x7fffu + ((ua >> 16) & 1u);
  return (unsigned short)(ua >> 16);
}
__device__ __forceinline__ float bf16_to_f(unsigned short u) {
  return __uint_as_float(((unsigned int)u) << 16);
}
__device__ __forceinline__ v8f wmma_bf16(v8u a, v8u b, v8f c) {
  return __builtin_amdgcn_wmma_f32_16x16x32_bf16(
      false, __builtin_bit_cast(v16bf, a),
      false, __builtin_bit_cast(v16bf, b),
      (short)0, c, false, false);
}
// A fragment (16x32 bf16): lane m = L&15; pair v holds K = kofs(v), kofs(v)+1
__device__ __forceinline__ int afrag_k(int lane, int v) {
  return ((lane >= 16) ? 8 : 0) + ((v >= 4) ? 16 : 0) + ((v & 3) << 1);
}
// B fragment (32x16 bf16): lane n = L&15; pair v holds K = kofs(v), kofs(v)+1
__device__ __forceinline__ int bfrag_k(int lane, int v) {
  return ((lane >= 16) ? 8 : 0) + ((v >= 4) ? 16 : 0) + ((v & 3) << 1);
}
__device__ __forceinline__ float rowmax16(float x) {
  x = fmaxf(x, __shfl_xor(x, 8, 32));
  x = fmaxf(x, __shfl_xor(x, 4, 32));
  x = fmaxf(x, __shfl_xor(x, 2, 32));
  x = fmaxf(x, __shfl_xor(x, 1, 32));
  return x;
}
__device__ __forceinline__ float rowsum16(float x) {
  x += __shfl_xor(x, 8, 32);
  x += __shfl_xor(x, 4, 32);
  x += __shfl_xor(x, 2, 32);
  x += __shfl_xor(x, 1, 32);
  return x;
}

// ---------- log-bucket table: tab[i] = clip(bucket(i-1023)+256, 0, 511) ----------
__global__ void bucket_kernel(int* __restrict__ tab) {
  int i = blockIdx.x * blockDim.x + threadIdx.x;
  if (i >= 2047) return;
  int rel = i - 1023;
  const float mid = 128.0f;                 // BUCKETS/2
  float sgn = (rel > 0) ? 1.f : ((rel < 0) ? -1.f : 0.f);
  float abs_pos = (rel < 128 && rel > -128) ? (mid - 1.0f) : fabsf((float)rel);
  float log_pos = ceilf(__logf(abs_pos / mid) / __logf(511.0f / mid) * (mid - 1.0f)) + mid;
  float bucket = (abs_pos <= mid) ? (float)rel : log_pos * sgn;
  int idx = (int)bucket + 256;
  idx = min(max(idx, 0), 511);
  tab[i] = idx;
}

// ---------- f32 GEMM  Y = X*W^T + b  -> bf16 head-major ----------
// X:[M][1024] f32, W:[1024][1024] f32, out:[b*16+h][RPB][64] bf16
// Block tile 128(M) x 64(N); each wave computes 32x64 (2 A-frags x 4 B-frags).
template <int RPB>   // rows per batch: 1024 (hidden) or 512 (rel_emb)
__global__ __launch_bounds__(128) void proj_kernel(
    const float* __restrict__ X, const float* __restrict__ W,
    const float* __restrict__ bias, unsigned short* __restrict__ out) {
  __shared__ unsigned int A_lds[128][16];  // 128 rows x 32 k (bf16 pairs)
  __shared__ unsigned int B_lds[64][16];
  const int K = 1024;
  const int mbase = blockIdx.x * 128;
  const int nbase = blockIdx.y * 64;
  const int t = threadIdx.x;
  const int wave = t >> 5, lane = t & 31;

  v8f acc[2][4] = {};
  float bias_n[4];
#pragma unroll
  for (int nf = 0; nf < 4; ++nf) bias_n[nf] = bias[nbase + nf * 16 + (lane & 15)];

  for (int k0 = 0; k0 < K; k0 += 32) {
    // prefetch next chunk's rows (one cacheline each; lowers to global_prefetch_b8)
    if (k0 + 32 < K) {
      __builtin_prefetch(&X[(size_t)(mbase + t) * K + k0 + 32], 0, 3);
      if (t < 64) __builtin_prefetch(&W[(size_t)(nbase + t) * K + k0 + 32], 0, 3);
    }
    // cooperative f32 -> bf16 staging
    for (int p = t; p < 2048; p += 128) {
      int row = p >> 4, kp = p & 15;
      float2 xa = *(const float2*)&X[(size_t)(mbase + row) * K + k0 + kp * 2];
      A_lds[row][kp] = pack_bf16(xa.x, xa.y);
    }
    for (int p = t; p < 1024; p += 128) {
      int row = p >> 4, kp = p & 15;
      float2 xb = *(const float2*)&W[(size_t)(nbase + row) * K + k0 + kp * 2];
      B_lds[row][kp] = pack_bf16(xb.x, xb.y);
    }
    __syncthreads();
    v8u a[2];
#pragma unroll
    for (int mf = 0; mf < 2; ++mf) {
      const int am = wave * 32 + mf * 16 + (lane & 15);
#pragma unroll
      for (int v = 0; v < 8; ++v) a[mf][v] = A_lds[am][afrag_k(lane, v) >> 1];
    }
#pragma unroll
    for (int nf = 0; nf < 4; ++nf) {
      v8u b;
      const int bn = nf * 16 + (lane & 15);
#pragma unroll
      for (int v = 0; v < 8; ++v) b[v] = B_lds[bn][bfrag_k(lane, v) >> 1];
#pragma unroll
      for (int mf = 0; mf < 2; ++mf) acc[mf][nf] = wmma_bf16(a[mf], b, acc[mf][nf]);
    }
    __syncthreads();
  }
  const int rowoff = ((lane >= 16) ? 8 : 0);
#pragma unroll
  for (int nf = 0; nf < 4; ++nf) {
    int n = nbase + nf * 16 + (lane & 15);
    int h = n >> 6, dd = n & 63;
#pragma unroll
    for (int mf = 0; mf < 2; ++mf) {
#pragma unroll
      for (int r = 0; r < 8; ++r) {
        int row = mbase + wave * 32 + mf * 16 + rowoff + r;
        int s = row & (RPB - 1);           // RPB is power of two -> shift/mask
        int bh = (row / RPB) * 16 + h;     // folds to shift at compile time
        out[((size_t)bh * RPB + s) * 64 + dd] =
            f_to_bf16(acc[mf][nf][r] + bias_n[nf]);
      }
    }
  }
}

// ---------- bf16 batched GEMM: out[bh][m][n] = scale * A[bh][m][:] . B[h][n][:] ----------
__global__ __launch_bounds__(128) void posgemm_kernel(
    const unsigned short* __restrict__ A,   // [32][1024][64] bf16
    const unsigned short* __restrict__ Bm,  // [16][512][64]  bf16
    unsigned short* __restrict__ out,       // [32][1024][512] bf16
    float scale) {
  const int t = threadIdx.x;
  const int wave = t >> 5, lane = t & 31;
  const int mbase = blockIdx.x * 64 + wave * 16;
  const int nbase = blockIdx.y * 64;
  const int bh = blockIdx.z, h = bh & 15;
  const uint32_t* A32 = (const uint32_t*)(A + (size_t)bh * 1024 * 64);
  const uint32_t* B32 = (const uint32_t*)(Bm + (size_t)h * 512 * 64);
  v8f acc[4] = {};
  const int am = mbase + (lane & 15);
#pragma unroll
  for (int j = 0; j < 2; ++j) {
    v8u a;
#pragma unroll
    for (int v = 0; v < 8; ++v)
      a[v] = A32[am * 32 + ((j * 32 + afrag_k(lane, v)) >> 1)];
#pragma unroll
    for (int nf = 0; nf < 4; ++nf) {
      const int bn = nbase + nf * 16 + (lane & 15);
      v8u b;
#pragma unroll
      for (int v = 0; v < 8; ++v)
        b[v] = B32[bn * 32 + ((j * 32 + bfrag_k(lane, v)) >> 1)];
      acc[nf] = wmma_bf16(a, b, acc[nf]);
    }
  }
  const int rowoff = (lane >= 16) ? 8 : 0;
#pragma unroll
  for (int nf = 0; nf < 4; ++nf) {
    int n = nbase + nf * 16 + (lane & 15);
#pragma unroll
    for (int r = 0; r < 8; ++r) {
      int m = mbase + rowoff + r;
      out[((size_t)bh * 1024 + m) * 512 + n] = f_to_bf16(acc[nf][r] * scale);
    }
  }
}

// ---------- fused flash attention with disentangled bias gathers ----------
__global__ __launch_bounds__(128) void attn_kernel(
    const unsigned short* __restrict__ Q,    // [32][1024][64] bf16
    const unsigned short* __restrict__ Kh,
    const unsigned short* __restrict__ Vh,
    const unsigned short* __restrict__ C2P,  // [32][1024][512] bf16 (pre-scaled)
    const unsigned short* __restrict__ P2C,
    const int* __restrict__ tab,
    const int* __restrict__ mask,            // [2][1024][1024]
    float* __restrict__ outp,                // [2][1024][1024] f32
    float scale) {
  __shared__ unsigned int  Kl[32 * 32];        // K chunk: 32 rows x 64 d (bf16 pairs)
  __shared__ unsigned short Vt[64][32];        // V chunk transposed: [d][k]
  __shared__ unsigned short Pst[4][16][32];    // per-wave P staging (C->A layout fix)

  const int S = 1024;
  const int t = threadIdx.x, wave = t >> 5, lane = t & 31;
  const int bh = blockIdx.y, b = bh >> 4, h = bh & 15;
  const int q0 = blockIdx.x * 64 + wave * 16;
  const int rowoff = (lane >= 16) ? 8 : 0;
  const int q_s0 = q0 + rowoff;

  // load Q fragments once (bf16 pairs contiguous in [s][d])
  const uint32_t* Q32 = (const uint32_t*)(Q + (size_t)bh * S * 64);
  const int qm = q0 + (lane & 15);
  v8u qa[2];
#pragma unroll
  for (int j = 0; j < 2; ++j)
#pragma unroll
    for (int v = 0; v < 8; ++v)
      qa[j][v] = Q32[qm * 32 + ((j * 32 + afrag_k(lane, v)) >> 1)];

  v8f O[4] = {};
  float m_i[8], l_i[8];
#pragma unroll
  for (int r = 0; r < 8; ++r) { m_i[r] = -3.0e38f; l_i[r] = 0.f; }

  const unsigned short* Kbase = Kh + (size_t)bh * S * 64;
  const unsigned short* Vbase = Vh + (size_t)bh * S * 64;
  const uint32_t* Pw32 = (const uint32_t*)&Pst[wave][0][0];

  for (int kc = 0; kc < S; kc += 32) {
    __syncthreads();
    {
      // K chunk: async LDS DMA (cache -> LDS, tracked with ASYNCcnt; no VGPR hop)
      const unsigned short* Ksrc = Kbase + (size_t)kc * 64;
#pragma unroll
      for (int it = 0; it < 2; ++it) {
        unsigned lds_off = (unsigned)(uintptr_t)(&Kl[it * 512 + t * 4]);
        uint64_t gaddr = (uint64_t)(uintptr_t)Ksrc + (size_t)(it * 2048 + t * 16);
        asm volatile("global_load_async_to_lds_b128 %0, %1, off"
                     :: "v"(lds_off), "v"(gaddr) : "memory");
      }
      // V chunk: staged + transposed manually (layout change in flight)
      const uint32_t* Vsrc = (const uint32_t*)(Vbase + (size_t)kc * 64);
      for (int p = t; p < 1024; p += 128) {
        uint32_t w2 = Vsrc[p];
        int krow = p >> 5, d2 = (p & 31) << 1;
        Vt[d2][krow]     = (unsigned short)(w2 & 0xffffu);
        Vt[d2 + 1][krow] = (unsigned short)(w2 >> 16);
      }
      asm volatile("s_wait_asynccnt 0" ::: "memory");
    }
    __syncthreads();

    // S chunk (16 q-rows x 32 k-cols) = Q(16x64) . K^T
    v8f sfr[2] = {};
#pragma unroll
    for (int nsub = 0; nsub < 2; ++nsub) {
      const int krow = nsub * 16 + (lane & 15);
#pragma unroll
      for (int j = 0; j < 2; ++j) {
        v8u bfr;
#pragma unroll
        for (int v = 0; v < 8; ++v)
          bfr[v] = Kl[krow * 32 + ((j * 32 + bfrag_k(lane, v)) >> 1)];
        sfr[nsub] = wmma_bf16(qa[j], bfr, sfr[nsub]);
      }
    }

    // scale + relative-position gathers + mask
#pragma unroll
    for (int nsub = 0; nsub < 2; ++nsub) {
      const int k_s = kc + nsub * 16 + (lane & 15);
#pragma unroll
      for (int r = 0; r < 8; ++r) {
        const int qs = q_s0 + r;
        const int rel = qs - k_s;
        const int ic = tab[rel + 1023];
        const int ip = tab[1023 - rel];
        float v = sfr[nsub][r] * scale
                + bf16_to_f(C2P[((size_t)bh * S + qs) * 512 + ic])
                + bf16_to_f(P2C[((size_t)bh * S + k_s) * 512 + ip]);
        if (mask[((size_t)b * S + qs) * S + k_s] == 0) v = -3.0e38f;
        sfr[nsub][r] = v;
      }
    }

    // online softmax (rows live per-VGPR; 16-lane groups hold 16 cols)
    float alpha[8];
#pragma unroll
    for (int r = 0; r < 8; ++r) {
      float rm = rowmax16(fmaxf(sfr[0][r], sfr[1][r]));
      float mn = fmaxf(m_i[r], rm);
      alpha[r] = __expf(m_i[r] - mn);
      m_i[r] = mn;
    }
#pragma unroll
    for (int r = 0; r < 8; ++r) {
      float p0 = __expf(sfr[0][r] - m_i[r]);
      float p1 = __expf(sfr[1][r] - m_i[r]);
      sfr[0][r] = p0; sfr[1][r] = p1;
      l_i[r] = l_i[r] * alpha[r] + rowsum16(p0 + p1);
    }
#pragma unroll
    for (int d = 0; d < 4; ++d)
#pragma unroll
      for (int r = 0; r < 8; ++r) O[d][r] *= alpha[r];

    // stage P (C layout -> bf16 LDS), wave-local: s_wait_dscnt then A-frag loads
#pragma unroll
    for (int nsub = 0; nsub < 2; ++nsub) {
      const int col = nsub * 16 + (lane & 15);
#pragma unroll
      for (int r = 0; r < 8; ++r)
        Pst[wave][rowoff + r][col] = f_to_bf16(sfr[nsub][r]);
    }
    asm volatile("s_wait_dscnt 0" ::: "memory");
    v8u pa;
    const int pm = lane & 15;
#pragma unroll
    for (int v = 0; v < 8; ++v)
      pa[v] = Pw32[pm * 16 + (afrag_k(lane, v) >> 1)];

    // O += P(16x32) . V(32x64)
#pragma unroll
    for (int d = 0; d < 4; ++d) {
      const int dcol = d * 16 + (lane & 15);
      const uint32_t* Vt32 = (const uint32_t*)&Vt[dcol][0];
      v8u bv;
#pragma unroll
      for (int v = 0; v < 8; ++v) bv[v] = Vt32[bfrag_k(lane, v) >> 1];
      O[d] = wmma_bf16(pa, bv, O[d]);
    }
  }

  // epilogue: normalize and scatter to [b][s][h*64+d] f32
#pragma unroll
  for (int r = 0; r < 8; ++r) {
    float inv = (l_i[r] > 0.f) ? 1.0f / l_i[r] : 0.f;
    const int qs = q_s0 + r;
#pragma unroll
    for (int d = 0; d < 4; ++d) {
      const int dcol = d * 16 + (lane & 15);
      outp[((size_t)b * S + qs) * 1024 + h * 64 + dcol] = O[d][r] * inv;
    }
  }
}

// ---------- host ----------
extern "C" void kernel_launch(void* const* d_in, const int* in_sizes, int n_in,
                              void* d_out, int out_size, void* d_ws, size_t ws_size,
                              hipStream_t stream) {
  const float* hidden   = (const float*)d_in[0];
  const int*   amask    = (const int*)d_in[1];
  const float* rel_emb  = (const float*)d_in[2];
  const float* q_w = (const float*)d_in[3];
  const float* q_b = (const float*)d_in[4];
  const float* k_w = (const float*)d_in[5];
  const float* k_b = (const float*)d_in[6];
  const float* v_w = (const float*)d_in[7];
  const float* v_b = (const float*)d_in[8];
  const float* pos_q_w = (const float*)d_in[9];
  const float* pos_q_b = (const float*)d_in[10];
  const float* pos_k_w = (const float*)d_in[11];
  const float* pos_k_b = (const float*)d_in[12];
  float* out = (float*)d_out;

  size_t off = 0;
  auto carve = [&](size_t bytes) -> void* {
    void* p = (char*)d_ws + off;
    off += (bytes + 255) & ~(size_t)255;
    return p;
  };
  unsigned short* qh   = (unsigned short*)carve((size_t)32 * 1024 * 64 * 2);
  unsigned short* kh   = (unsigned short*)carve((size_t)32 * 1024 * 64 * 2);
  unsigned short* vh   = (unsigned short*)carve((size_t)32 * 1024 * 64 * 2);
  unsigned short* posk = (unsigned short*)carve((size_t)16 * 512 * 64 * 2);
  unsigned short* posq = (unsigned short*)carve((size_t)16 * 512 * 64 * 2);
  unsigned short* c2p  = (unsigned short*)carve((size_t)32 * 1024 * 512 * 2);
  unsigned short* p2c  = (unsigned short*)carve((size_t)32 * 1024 * 512 * 2);
  int* tab = (int*)carve(2048 * sizeof(int));

  const float scale = 1.0f / sqrtf(64.0f * 3.0f);

  bucket_kernel<<<dim3(8), dim3(256), 0, stream>>>(tab);

  // Q/K/V projections: M = 2048 rows -> 16 blocks of 128
  proj_kernel<1024><<<dim3(16, 16), dim3(128), 0, stream>>>(hidden, q_w, q_b, qh);
  proj_kernel<1024><<<dim3(16, 16), dim3(128), 0, stream>>>(hidden, k_w, k_b, kh);
  proj_kernel<1024><<<dim3(16, 16), dim3(128), 0, stream>>>(hidden, v_w, v_b, vh);
  // pos projections: M = 512 rows -> 4 blocks of 128
  proj_kernel<512><<<dim3(4, 16), dim3(128), 0, stream>>>(rel_emb, pos_k_w, pos_k_b, posk);
  proj_kernel<512><<<dim3(4, 16), dim3(128), 0, stream>>>(rel_emb, pos_q_w, pos_q_b, posq);

  // c2p = scale * q . pos_k^T ; p2c = scale * k . pos_q^T
  posgemm_kernel<<<dim3(16, 8, 32), dim3(128), 0, stream>>>(qh, posk, c2p, scale);
  posgemm_kernel<<<dim3(16, 8, 32), dim3(128), 0, stream>>>(kh, posq, p2c, scale);

  // fused attention
  attn_kernel<<<dim3(16, 32), dim3(128), 0, stream>>>(
      qh, kh, vh, c2p, p2c, tab, amask, out, scale);
}
